// GNNModel_40767829573778
// MI455X (gfx1250) — matrix operand order repaired and checked by reading, not compiled, not run on hardware
//
#include <hip/hip_runtime.h>
#include <hip/hip_bf16.h>

// ---------------------------------------------------------------------------
// GraphSAGE forward on MI455X (gfx1250, wave32).
//  - Edge/pool scatter: coalesced float4 gathers + native global_atomic_add_f32
//    (inline asm; destination arrays are L2-resident: ~60 MB << 192 MB L2).
//  - Dense layers: v_wmma_f32_16x16x32_bf16, fp32 accumulate. Weights are
//    pre-packed once per call into WMMA B-fragment layout (bf16) so each
//    fragment is a single 32B vector load instead of 16 strided scalar loads.
// ---------------------------------------------------------------------------

typedef __attribute__((ext_vector_type(16))) __bf16 v16bf;
typedef __attribute__((ext_vector_type(8)))  float  v8f;

constexpr int N_NODES = 50000;
constexpr int N_EDGES = 1000000;
constexpr int DIM     = 64;    // D == H == 64
constexpr int OUT_D   = 16;
constexpr int N_GR    = 512;

// ---------------------------------------------------------------------------
// helpers
// ---------------------------------------------------------------------------
__device__ __forceinline__ void atomic_add_f32(float* p, float v) {
  // Non-returning fp32 atomic (STOREcnt-tracked); guaranteed native L2 atomic.
  asm volatile("global_atomic_add_f32 %0, %1, off" :: "v"(p), "v"(v) : "memory");
}

// A fragment (16x32 bf16, MxK): lane = m + 16*half; per ISA table the lane
// holds K = kb + 8*half + {0..7}  and  K = kb + 16 + 8*half + {0..7}.
__device__ __forceinline__ v16bf load_fragA(const float* __restrict__ A,
                                            int row, int kb, int half) {
  const float* p = A + row * DIM + kb + 8 * half;
  float t[16];
  *(float4*)(t + 0)  = *(const float4*)(p + 0);
  *(float4*)(t + 4)  = *(const float4*)(p + 4);
  *(float4*)(t + 8)  = *(const float4*)(p + 16);
  *(float4*)(t + 12) = *(const float4*)(p + 20);
  v16bf a;
#pragma unroll
  for (int i = 0; i < 16; ++i) a[i] = (__bf16)t[i];
  return a;
}

// ---------------------------------------------------------------------------
// kernels
// ---------------------------------------------------------------------------
__global__ void zero_kernel(float* __restrict__ p, int n) {
  int i = blockIdx.x * 256 + threadIdx.x;
  if (i < n) p[i] = 0.0f;
}

// Pack W [64][ldw] (row-major fp32) into bf16 WMMA B fragments.
// Fragment f = kb2*nt_tiles + nt; per lane 16 bf16: element j =
// W[(kb2*32 + 8*(lane/16) + (j<8 ? j : j+8))*ldw + nt*16 + lane%16].
__global__ void pack_w_kernel(const float* __restrict__ W, int ldw,
                              int nt_tiles, __bf16* __restrict__ out) {
  int t = blockIdx.x * 64 + threadIdx.x;
  if (t >= nt_tiles * 2 * 32) return;
  int lane = t & 31;
  int f    = t >> 5;
  int kb2  = f / nt_tiles;
  int nt   = f % nt_tiles;
  int half = lane >> 4;
  int ncol = nt * 16 + (lane & 15);
  int kbase = kb2 * 32 + 8 * half;
  __bf16* dst = out + ((size_t)f * 32 + lane) * 16;
#pragma unroll
  for (int j = 0; j < 16; ++j) dst[j] = (__bf16)W[(kbase + (j < 8 ? j : j + 8)) * ldw + ncol];
}

// One edge per 16 threads; each thread moves one float4 of the 64-float row.
__global__ void edge_scatter_kernel(const float* __restrict__ feat,
                                    const int* __restrict__ src,
                                    const int* __restrict__ dst,
                                    float* __restrict__ agg) {
  int tid = blockIdx.x * 256 + threadIdx.x;   // < E*16
  int e = tid >> 4;
  int q = tid & 15;
  int s = src[e];
  int d = dst[e];
  float4 v = ((const float4*)(feat + s * DIM))[q];
  float* base = agg + d * DIM + q * 4;
  atomic_add_f32(base + 0, v.x);
  atomic_add_f32(base + 1, v.y);
  atomic_add_f32(base + 2, v.z);
  atomic_add_f32(base + 3, v.w);
}

// Global add-pool: one node per 16 threads.
__global__ void pool_kernel(const float* __restrict__ h,
                            const int* __restrict__ batch,
                            float* __restrict__ g) {
  int tid = blockIdx.x * 256 + threadIdx.x;   // < N*16
  int i = tid >> 4;
  int q = tid & 15;
  int b = batch[i];
  float4 v = ((const float4*)(h + i * DIM))[q];
  float* base = g + b * DIM + q * 4;
  atomic_add_f32(base + 0, v.x);
  atomic_add_f32(base + 1, v.y);
  atomic_add_f32(base + 2, v.z);
  atomic_add_f32(base + 3, v.w);
}

// Fused SAGE layer: out = relu(A1 @ Wl + A2 @ Wr + bias).
// One wave per 16-row M tile; the wave owns all four 16-col N tiles
// (4 accumulators, 16 WMMAs). A fragments loaded once per K-half per matrix;
// B fragments are single 32B loads from the packed buffers.
__global__ __launch_bounds__(32) void sage_gemm_kernel(
    const float* __restrict__ A1,   // [N,64] neighbor aggregate
    const float* __restrict__ A2,   // [N,64] root features
    const v16bf* __restrict__ Bl,   // packed Wl fragments [8][32]
    const v16bf* __restrict__ Br,   // packed Wr fragments [8][32]
    const float* __restrict__ bias, // [64]
    float* __restrict__ out)        // [N,64]
{
  const int lane = threadIdx.x & 31;
  const int mt   = blockIdx.x;             // M tile
  const int mrow = mt * 16 + (lane & 15);
  const int half = lane >> 4;

  v8f c[4];
#pragma unroll
  for (int nt = 0; nt < 4; ++nt) {
    float bv = bias[nt * 16 + (lane & 15)];
#pragma unroll
    for (int r = 0; r < 8; ++r) c[nt][r] = bv;
  }

#pragma unroll
  for (int kb2 = 0; kb2 < 2; ++kb2) {
    const int kb = kb2 * 32;
    v16bf a1 = load_fragA(A1, mrow, kb, half);
    v16bf a2 = load_fragA(A2, mrow, kb, half);
#pragma unroll
    for (int nt = 0; nt < 4; ++nt) {
      v16bf bl = Bl[(kb2 * 4 + nt) * 32 + lane];
      c[nt] = __builtin_amdgcn_wmma_f32_16x16x32_bf16(false, a1, false, bl,
                                                      (short)0, c[nt], false, false);
      v16bf br = Br[(kb2 * 4 + nt) * 32 + lane];
      c[nt] = __builtin_amdgcn_wmma_f32_16x16x32_bf16(false, a2, false, br,
                                                      (short)0, c[nt], false, false);
    }
  }

  // D element r -> row mt*16 + r + 8*half, col nt*16 + lane%16.
#pragma unroll
  for (int nt = 0; nt < 4; ++nt) {
    float* op = out + (mt * 16 + 8 * half) * DIM + nt * 16 + (lane & 15);
#pragma unroll
    for (int r = 0; r < 8; ++r) op[r * DIM] = fmaxf(c[nt][r], 0.0f);
  }
}

// Head: out[512,16] = g[512,64] @ Wo[64,16] + bo. One wave per 16 rows;
// the 16-wide N dimension is exactly one WMMA tile.
__global__ __launch_bounds__(32) void head_kernel(
    const float* __restrict__ g, const v16bf* __restrict__ Bo,
    const float* __restrict__ bo, float* __restrict__ out) {
  const int lane = threadIdx.x & 31;
  const int mt   = blockIdx.x;
  const int mrow = mt * 16 + (lane & 15);
  const int half = lane >> 4;
  const int ncol = lane & 15;

  const float bv = bo[ncol];
  v8f c;
#pragma unroll
  for (int r = 0; r < 8; ++r) c[r] = bv;

#pragma unroll
  for (int kb2 = 0; kb2 < 2; ++kb2) {
    v16bf a = load_fragA(g, mrow, kb2 * 32, half);
    v16bf b = Bo[kb2 * 32 + lane];
    c = __builtin_amdgcn_wmma_f32_16x16x32_bf16(false, a, false, b,
                                                (short)0, c, false, false);
  }

  float* op = out + (mt * 16 + 8 * half) * OUT_D + ncol;
#pragma unroll
  for (int r = 0; r < 8; ++r) op[r * OUT_D] = c[r];
}

// ---------------------------------------------------------------------------
// launcher
// ---------------------------------------------------------------------------
extern "C" void kernel_launch(void* const* d_in, const int* in_sizes, int n_in,
                              void* d_out, int out_size, void* d_ws, size_t ws_size,
                              hipStream_t stream) {
  (void)in_sizes; (void)n_in; (void)out_size; (void)ws_size;

  const float* x     = (const float*)d_in[0];
  const int*   ei    = (const int*)d_in[1];      // [2,E]
  const int*   src   = ei;
  const int*   dst   = ei + N_EDGES;
  const int*   batch = (const int*)d_in[2];
  const float* Wl1   = (const float*)d_in[3];
  const float* Wr1   = (const float*)d_in[4];
  const float* b1    = (const float*)d_in[5];
  const float* Wl2   = (const float*)d_in[6];
  const float* Wr2   = (const float*)d_in[7];
  const float* b2    = (const float*)d_in[8];
  const float* Wo    = (const float*)d_in[9];
  const float* bo    = (const float*)d_in[10];
  float*       out   = (float*)d_out;

  // workspace: agg | h1 | h2 | g | packed weights (bf16)
  float* agg = (float*)d_ws;
  float* h1  = agg + (size_t)N_NODES * DIM;
  float* h2  = h1  + (size_t)N_NODES * DIM;
  float* g   = h2  + (size_t)N_NODES * DIM;
  __bf16* pWl1 = (__bf16*)(g + (size_t)N_GR * DIM);
  __bf16* pWr1 = pWl1 + 8 * 32 * 16;             // 4096 bf16 per 64x64 matrix
  __bf16* pWl2 = pWr1 + 8 * 32 * 16;
  __bf16* pWr2 = pWl2 + 8 * 32 * 16;
  __bf16* pWo  = pWr2 + 8 * 32 * 16;             // 2*32*16 = 1024 bf16

  const int nodeFeat   = N_NODES * DIM;          // 3.2M
  const int zeroBlocks = (nodeFeat + 255) / 256;
  const int edgeBlocks = (N_EDGES * 16) / 256;   // 62500
  const int poolBlocks = (N_NODES * 16) / 256;   // 3125
  const int gemmBlocks = N_NODES / 16;           // 3125

  // ---- pack weights into WMMA fragment layout ----
  pack_w_kernel<<<4, 64, 0, stream>>>(Wl1, DIM, 4, pWl1);
  pack_w_kernel<<<4, 64, 0, stream>>>(Wr1, DIM, 4, pWr1);
  pack_w_kernel<<<4, 64, 0, stream>>>(Wl2, DIM, 4, pWl2);
  pack_w_kernel<<<4, 64, 0, stream>>>(Wr2, DIM, 4, pWr2);
  pack_w_kernel<<<1, 64, 0, stream>>>(Wo, OUT_D, 1, pWo);

  // ---- layer 1 ----
  zero_kernel<<<zeroBlocks, 256, 0, stream>>>(agg, nodeFeat);
  edge_scatter_kernel<<<edgeBlocks, 256, 0, stream>>>(x, src, dst, agg);
  sage_gemm_kernel<<<gemmBlocks, 32, 0, stream>>>(agg, x, (const v16bf*)pWl1,
                                                  (const v16bf*)pWr1, b1, h1);

  // ---- layer 2 ----
  zero_kernel<<<zeroBlocks, 256, 0, stream>>>(agg, nodeFeat);
  edge_scatter_kernel<<<edgeBlocks, 256, 0, stream>>>(h1, src, dst, agg);
  sage_gemm_kernel<<<gemmBlocks, 32, 0, stream>>>(agg, h1, (const v16bf*)pWl2,
                                                  (const v16bf*)pWr2, b2, h2);

  // ---- pool + head ----
  zero_kernel<<<(N_GR * DIM + 255) / 256, 256, 0, stream>>>(g, N_GR * DIM);
  pool_kernel<<<poolBlocks, 256, 0, stream>>>(h2, batch, g);
  head_kernel<<<N_GR / 16, 32, 0, stream>>>(g, (const v16bf*)pWo, bo, out);
}
